// GAT_13494787244282
// MI455X (gfx1250) — matrix operand order repaired and checked by reading, not compiled, not run on hardware
//
#include <hip/hip_runtime.h>
#include <math.h>

// ---------------- problem constants (match reference) ----------------
#define N_NODES 50000
#define N_EDGES 800000
#define IN_DIM  128
#define HID     64
#define HEADS   4
#define SLOPE   0.01f

// ---------------- vector types (gfx1250, wave32) ----------------
typedef __attribute__((ext_vector_type(16))) __bf16        v16bf;
typedef __attribute__((ext_vector_type(8)))  float         v8f;
typedef __attribute__((ext_vector_type(4)))  unsigned int  v4u;
typedef __attribute__((ext_vector_type(8)))  int           v8i;
typedef __attribute__((ext_vector_type(4)))  int           v4i;

union BFFrag { v16bf v; v4u q[2]; };   // 32B: 16 bf16 = two 16B quads

__device__ __forceinline__ unsigned short f2bf_bits(float f) {
    union { float f; unsigned u; } c; c.f = f;
    unsigned r = c.u + 0x7FFFu + ((c.u >> 16) & 1u);   // round-to-nearest-even
    return (unsigned short)(r >> 16);
}

// Monotone float atomic-max via int/uint atomics (init must be -inf).
__device__ __forceinline__ void atomicMaxFloat(float* addr, float val) {
    if (val >= 0.0f) atomicMax((int*)addr, __float_as_int(val));
    else             atomicMin((unsigned int*)addr, __float_as_uint(val));
}

// ---------------- fp32 -> bf16 bit conversion (elementwise) ---------------
__global__ void k_cvt_bf16(const float* __restrict__ in,
                           unsigned short* __restrict__ out, long long n) {
    const long long i = (long long)blockIdx.x * blockDim.x + threadIdx.x;
    if (i < n) out[i] = f2bf_bits(in[i]);
}

// ---------------- GEMM: C[M,Co] = Xbf[M,K] * Wbf[K,Co], f32 acc ------------
// Block: 8 waves = 128 rows x 64 cols of C. B slice (K x 64 bf16) staged in
// LDS by the Tensor Data Mover (wave 0 issues TENSOR_LOAD_TO_LDS, waits on
// TENSORcnt, then workgroup barrier). Each wave: 16x64 tile = 4 accumulators,
// A fragment reused 4x, 4 v_wmma_f32_16x16x32_bf16 per K-step of 32.
template <int K, int Co>
__global__ void k_gemm_wmma(const unsigned short* __restrict__ Xbf,
                            const unsigned short* __restrict__ Wbf,
                            float* __restrict__ C, int M) {
    __shared__ unsigned short sB[K * 64];

    const int wave = threadIdx.x >> 5;
    const int lane = threadIdx.x & 31;
    const int col0 = blockIdx.y << 6;            // 64-wide column group

    // ---- stage B slice (K x 64 bf16 @ Wbf[:, col0:col0+64]) via TDM ----
    if (wave == 0) {
        const unsigned long long gaddr =
            (unsigned long long)(uintptr_t)(Wbf + col0);
        const unsigned ldsa = (unsigned)(uintptr_t)(void*)&sB[0];
        v4u g0;
        g0.x = 1u;                                            // count=1 user D#
        g0.y = ldsa;                                          // lds_addr
        g0.z = (unsigned)(gaddr & 0xFFFFFFFFu);               // global_addr[31:0]
        g0.w = (unsigned)((gaddr >> 32) & 0x1FFFFFFu) | (2u << 30); // [56:32]+type=2
        v8i g1;
        g1[0] = (int)(1u << 16);                   // data_size=1 (2 bytes), no mcast
        g1[1] = (int)((unsigned)Co << 16);         // tensor_dim0[15:0] @ bits 63:48
        g1[2] = (int)(((unsigned)K & 0xFFFFu) << 16); // tensor_dim1[15:0] @ 95:80
        g1[3] = (int)(64u << 16);                  // tile_dim0=64 @ bits 127:112
        g1[4] = (int)(unsigned)K;                  // tile_dim1=K @ bits 143:128
        g1[5] = (int)(unsigned)Co;                 // tensor_dim0_stride[31:0]
        g1[6] = 0;
        g1[7] = 0;
        v4i g2 = {0, 0, 0, 0};
        v4i g3 = {0, 0, 0, 0};
#if __has_include(<hip/amd_detail/amd_gfx1250_TDM.h>)
        v8i g4 = {0, 0, 0, 0, 0, 0, 0, 0};
        __builtin_amdgcn_tensor_load_to_lds(g0, g1, g2, g3, g4, 0);
#else
        __builtin_amdgcn_tensor_load_to_lds(g0, g1, g2, g3, 0);
#endif
        __builtin_amdgcn_s_wait_tensorcnt(0);
    }
    __syncthreads();

    const int tilem = blockIdx.x * 8 + wave;
    if (tilem >= (M >> 4)) return;
    const int row0 = tilem << 4;

    v8f acc[4] = {};
    const int m  = lane & 15;
    const int kb = (lane >> 4) << 3;             // element offset 0 / 8
    const unsigned short* __restrict__ arow = Xbf + (size_t)(row0 + m) * K;

    for (int k0 = 0; k0 < K; k0 += 32) {
        BFFrag a;
        a.q[0] = *(const v4u*)(arow + k0 + kb);        // K = kb+0..7
        a.q[1] = *(const v4u*)(arow + k0 + kb + 16);   // K = kb+16..23
        #pragma unroll
        for (int nt = 0; nt < 4; ++nt) {
            BFFrag b;                                   // lane = N col, elems = K (kb+0..7, kb+16..23)
            const unsigned short* bcol = &sB[(k0 + kb) * 64 + (nt << 4) + m];
            #pragma unroll
            for (int s = 0; s < 8; ++s) {
                ((unsigned short*)&b)[s]     = bcol[s * 64];
                ((unsigned short*)&b)[s + 8] = bcol[(s + 16) * 64];
            }
            acc[nt] = __builtin_amdgcn_wmma_f32_16x16x32_bf16(
                          false, a.v, false, b.v, (short)0, acc[nt], false, false);
        }
    }

    // C/D layout: VGPR r -> M = r + (lane>=16 ? 8 : 0), N = lane&15
    const int mo = (lane >> 4) << 3;
    const int n  = lane & 15;
    #pragma unroll
    for (int nt = 0; nt < 4; ++nt)
        #pragma unroll
        for (int r = 0; r < 8; ++r)
            C[(size_t)(row0 + mo + r) * Co + col0 + (nt << 4) + n] = acc[nt][r];
}

// ---------------- per-node attention projections: el/er [N,H] --------------
__global__ void k_el_er(const float* __restrict__ ft,
                        const float* __restrict__ al,
                        const float* __restrict__ ar,
                        float* __restrict__ el, float* __restrict__ er,
                        int H, int D) {
    const int wid  = (blockIdx.x * blockDim.x + threadIdx.x) >> 5;
    const int lane = threadIdx.x & 31;
    if (wid >= N_NODES * H) return;
    const int n = wid / H, h = wid - n * H;
    const float* __restrict__ f   = ft + (size_t)n * H * D + (size_t)h * D;
    const float* __restrict__ alh = al + (size_t)h * D;
    const float* __restrict__ arh = ar + (size_t)h * D;
    float sl = 0.f, sr = 0.f;
    for (int d = lane; d < D; d += 32) { float v = f[d]; sl += v * alh[d]; sr += v * arh[d]; }
    #pragma unroll
    for (int off = 16; off > 0; off >>= 1) {
        sl += __shfl_xor(sl, off, 32);
        sr += __shfl_xor(sr, off, 32);
    }
    if (lane == 0) { el[wid] = sl; er[wid] = sr; }
}

// ---------------- init: nmax=-inf, nsum=0, agg=0 ---------------------------
__global__ void k_init(float* __restrict__ nmax, float* __restrict__ nsum,
                       float* __restrict__ agg, long long nh, long long total_agg) {
    const long long i = (long long)blockIdx.x * blockDim.x + threadIdx.x;
    if (i < nh) { nmax[i] = -__builtin_inff(); nsum[i] = 0.0f; }
    if (i < total_agg) agg[i] = 0.0f;
}

// ---------------- edge pass 1: logit = leaky_relu, segment max -------------
__global__ void k_edge_max(const int* __restrict__ src, const int* __restrict__ dst,
                           const float* __restrict__ el, const float* __restrict__ er,
                           float* __restrict__ logit, float* __restrict__ nmax, int H) {
    const long long i = (long long)blockIdx.x * blockDim.x + threadIdx.x;
    if (i >= (long long)N_EDGES * H) return;
    const int e = (int)(i / H), h = (int)(i - (long long)e * H);
    const int s = src[e], d = dst[e];
    float x = el[s * H + h] + er[d * H + h];
    x = (x > 0.0f) ? x : SLOPE * x;
    logit[i] = x;
    atomicMaxFloat(&nmax[d * H + h], x);
}

// ---------------- edge pass 2: ex = exp(logit - max), segment sum ----------
__global__ void k_edge_expsum(const int* __restrict__ dst,
                              const float* __restrict__ nmax,
                              float* __restrict__ exv,      // in: logit, out: ex
                              float* __restrict__ nsum, int H) {
    const long long i = (long long)blockIdx.x * blockDim.x + threadIdx.x;
    if (i >= (long long)N_EDGES * H) return;
    const int e = (int)(i / H), h = (int)(i - (long long)e * H);
    const int d = dst[e];
    const float ex = expf(exv[i] - nmax[d * H + h]);
    exv[i] = ex;
    atomicAdd(&nsum[d * H + h], ex);
}

// ---------------- edge pass 3: scatter msg = ft[src]*a into agg[dst] -------
// one wave per (edge, head); lane covers D=64 in 2 steps
__global__ void k_edge_scatter(const int* __restrict__ src, const int* __restrict__ dst,
                               const float* __restrict__ exv, const float* __restrict__ nsum,
                               const float* __restrict__ ft, float* __restrict__ agg,
                               int H, int D) {
    const int wid  = (blockIdx.x * blockDim.x + threadIdx.x) >> 5;
    const int lane = threadIdx.x & 31;
    if (wid >= N_EDGES * H) return;
    const int e = wid / H, h = wid - e * H;
    const int s = src[e], d = dst[e];
    const float* __restrict__ fs = ft + (size_t)s * H * D + (size_t)h * D;
    __builtin_prefetch(fs, 0, 0);                       // global_prefetch_b8
    const float a = exv[wid] / nsum[d * H + h];
    float* __restrict__ od = agg + (size_t)d * H * D + (size_t)h * D;
    for (int t = lane; t < D; t += 32)
        atomicAdd(&od[t], fs[t] * a);
}

// ---------------- out = elu(in + bias), fp32 -------------------------------
__global__ void k_elu_bias(const float* __restrict__ in, const float* __restrict__ bias,
                           float* __restrict__ out, int C, long long total) {
    const long long i = (long long)blockIdx.x * blockDim.x + threadIdx.x;
    if (i >= total) return;
    const int c = (int)(i % C);
    const float x = in[i] + bias[c];
    out[i] = (x > 0.0f) ? x : expm1f(x);
}

// ---------------- outbf = bf16(elu(in + bias)) (feeds layer-2 GEMM) --------
__global__ void k_elu_bias_bf16(const float* __restrict__ in, const float* __restrict__ bias,
                                unsigned short* __restrict__ outbf, int C, long long total) {
    const long long i = (long long)blockIdx.x * blockDim.x + threadIdx.x;
    if (i >= total) return;
    const int c = (int)(i % C);
    const float x = in[i] + bias[c];
    outbf[i] = f2bf_bits((x > 0.0f) ? x : expm1f(x));
}

// ---------------------------------------------------------------------------
static inline int cdiv_ll(long long a, long long b) { return (int)((a + b - 1) / b); }

extern "C" void kernel_launch(void* const* d_in, const int* in_sizes, int n_in,
                              void* d_out, int out_size, void* d_ws, size_t ws_size,
                              hipStream_t stream) {
    (void)in_sizes; (void)n_in; (void)out_size; (void)ws_size;

    const float* x   = (const float*)d_in[0];
    const int*   src = (const int*)  d_in[1];
    const int*   dst = (const int*)  d_in[2];
    const float* W1  = (const float*)d_in[3];
    const float* al1 = (const float*)d_in[4];
    const float* ar1 = (const float*)d_in[5];
    const float* b1  = (const float*)d_in[6];
    const float* W2  = (const float*)d_in[7];
    const float* al2 = (const float*)d_in[8];
    const float* ar2 = (const float*)d_in[9];
    const float* b2  = (const float*)d_in[10];
    float* out = (float*)d_out;

    // -------- workspace layout --------
    float* ws = (float*)d_ws;
    size_t o = 0;
    float* ft1   = ws + o; o += (size_t)N_NODES * HEADS * HID;   // 12.8M f
    float* agg1  = ws + o; o += (size_t)N_NODES * HEADS * HID;   // 12.8M f
    float* el1   = ws + o; o += (size_t)N_NODES * HEADS;
    float* er1   = ws + o; o += (size_t)N_NODES * HEADS;
    float* nmax1 = ws + o; o += (size_t)N_NODES * HEADS;
    float* nsum1 = ws + o; o += (size_t)N_NODES * HEADS;
    float* exv1  = ws + o; o += (size_t)N_EDGES * HEADS;
    unsigned short* Xbf  = (unsigned short*)(ws + o); o += (size_t)N_NODES * IN_DIM / 2;        // bf16
    unsigned short* H1bf = (unsigned short*)(ws + o); o += (size_t)N_NODES * HEADS * HID / 2;   // bf16
    unsigned short* W1bf = (unsigned short*)(ws + o); o += (size_t)IN_DIM * HEADS * HID / 2;
    unsigned short* W2bf = (unsigned short*)(ws + o); o += (size_t)HEADS * HID * HID / 2;
    // layer-2 scratch aliases ft1 (dead after layer-1 scatter)
    float* ft2   = ft1;
    float* agg2  = ft2   + (size_t)N_NODES * HID;
    float* el2   = agg2  + (size_t)N_NODES * HID;
    float* er2   = el2   + N_NODES;
    float* nmax2 = er2   + N_NODES;
    float* nsum2 = nmax2 + N_NODES;
    float* exv2  = nsum2 + N_NODES;

    const int B = 256;
    const int rowBlocks = cdiv_ll(N_NODES / 16, 8);   // 391 blocks of 128 rows

    // ================= layer 1 (H=4, D=64, K=128, Co=256) =================
    {
        k_cvt_bf16<<<cdiv_ll((long long)N_NODES * IN_DIM, B), B, 0, stream>>>(
            x, Xbf, (long long)N_NODES * IN_DIM);
        k_cvt_bf16<<<cdiv_ll((long long)IN_DIM * HEADS * HID, B), B, 0, stream>>>(
            W1, W1bf, (long long)IN_DIM * HEADS * HID);

        k_gemm_wmma<IN_DIM, HEADS * HID>
            <<<dim3(rowBlocks, (HEADS * HID) / 64), B, 0, stream>>>(Xbf, W1bf, ft1, N_NODES);

        const long long nhw = (long long)N_NODES * HEADS;
        k_el_er<<<cdiv_ll(nhw * 32, B), B, 0, stream>>>(ft1, al1, ar1, el1, er1, HEADS, HID);

        const long long tot1 = (long long)N_NODES * HEADS * HID;
        k_init<<<cdiv_ll(tot1, B), B, 0, stream>>>(nmax1, nsum1, agg1, nhw, tot1);

        const long long eh = (long long)N_EDGES * HEADS;
        k_edge_max<<<cdiv_ll(eh, B), B, 0, stream>>>(src, dst, el1, er1, exv1, nmax1, HEADS);
        k_edge_expsum<<<cdiv_ll(eh, B), B, 0, stream>>>(dst, nmax1, exv1, nsum1, HEADS);
        k_edge_scatter<<<cdiv_ll(eh * 32, B), B, 0, stream>>>(src, dst, exv1, nsum1, ft1, agg1, HEADS, HID);

        k_elu_bias_bf16<<<cdiv_ll(tot1, B), B, 0, stream>>>(agg1, b1, H1bf, HEADS * HID, tot1);
    }

    // ================= layer 2 (H=1, D=64, K=256, Co=64) ==================
    {
        k_cvt_bf16<<<cdiv_ll((long long)HEADS * HID * HID, B), B, 0, stream>>>(
            W2, W2bf, (long long)HEADS * HID * HID);

        k_gemm_wmma<HEADS * HID, HID>
            <<<dim3(rowBlocks, 1), B, 0, stream>>>(H1bf, W2bf, ft2, N_NODES);

        const long long nhw = (long long)N_NODES;
        k_el_er<<<cdiv_ll(nhw * 32, B), B, 0, stream>>>(ft2, al2, ar2, el2, er2, 1, HID);

        const long long tot2 = (long long)N_NODES * HID;
        k_init<<<cdiv_ll(tot2, B), B, 0, stream>>>(nmax2, nsum2, agg2, nhw, tot2);

        const long long eh = (long long)N_EDGES;
        k_edge_max<<<cdiv_ll(eh, B), B, 0, stream>>>(src, dst, el2, er2, exv2, nmax2, 1);
        k_edge_expsum<<<cdiv_ll(eh, B), B, 0, stream>>>(dst, nmax2, exv2, nsum2, 1);
        k_edge_scatter<<<cdiv_ll(eh * 32, B), B, 0, stream>>>(src, dst, exv2, nsum2, ft2, agg2, 1, HID);

        k_elu_bias<<<cdiv_ll(tot2, B), B, 0, stream>>>(agg2, b2, out, HID, tot2);
    }
}
